// GeometryAwareDescriptorMatcher_8272107012372
// MI455X (gfx1250) — matrix operand order repaired and checked by reading, not compiled, not run on hardware
//
#include <hip/hip_runtime.h>
#include <hip/hip_bf16.h>

// GeometryAwareDescriptorMatcher for MI455X (gfx1250, wave32, WMMA).
// Fused cdist + top-10 so the 256MB distance matrix never hits HBM.
//   Kernel A: column norms |desc2_j|^2
//   Kernel B: fused WMMA-GEMM (f16 in, f32 acc) + per-row running top-10
//   Kernel C: FGINN ratio test (faithful to reference's candidates_xy[0:1]
//             broadcast of ROW 0's candidates -> needs global results first)

typedef __attribute__((ext_vector_type(16))) _Float16 v16h;
typedef __attribute__((ext_vector_type(8)))  _Float16 v8h;
typedef __attribute__((ext_vector_type(8)))  float    v8f;

#define DDIM 128
#define TOPK 10
#define FLT_BIG 3.4e38f

// ---------------- Kernel A: squared norms of desc2 rows ----------------
__global__ void norms2_kernel(const float* __restrict__ desc,
                              float* __restrict__ n2, int B) {
  int i = blockIdx.x * blockDim.x + threadIdx.x;
  if (i >= B) return;
  const float4* p = (const float4*)(desc + (size_t)i * DDIM);
  float s = 0.f;
#pragma unroll
  for (int q = 0; q < DDIM / 4; q++) {
    float4 v = p[q];
    s += v.x * v.x + v.y * v.y + v.z * v.z + v.w * v.w;
  }
  n2[i] = s;
}

// ---------------- Kernel B: fused GEMM + top-10 ----------------
// 256 threads = 8 wave32; WG owns 16 rows of desc1; loops over desc2 in
// 128-column tiles. Wave w computes the 16x16 tile for columns w*16..w*16+15.
__global__ __launch_bounds__(256) void match_topk_kernel(
    const float* __restrict__ desc1, const float* __restrict__ desc2,
    const float* __restrict__ n2g, float* __restrict__ valsOut,
    int* __restrict__ idxOut, int B2) {
  // LDS plan (45,120 B total; merge buffer aliases the dead B tile):
  //   [    0,  4096) A tile   16x128 f16
  //   [ 4096,  4160) row norms n1[16]
  //   [ 4160, 12352) dist tile 16x128 f32
  //   [12352, 45120) B tile  128x128 f16   (aliased after loop by merge bufs)
  __shared__ __align__(32) unsigned char smem[45120];
  _Float16(*At)[DDIM] = (_Float16(*)[DDIM])(smem);
  float* shN1 = (float*)(smem + 4096);
  float(*dist)[DDIM] = (float(*)[DDIM])(smem + 4160);
  _Float16(*Bt)[DDIM] = (_Float16(*)[DDIM])(smem + 12352);
  float* mvals = (float*)(smem + 12352);   // alias: 16 rows * 160 cand
  int* midx = (int*)(smem + 22592);        // alias: 16 rows * 160 cand

  const int tid = threadIdx.x;
  const int lane = tid & 31;
  const int wave = tid >> 5;  // 0..7
  const int r0 = blockIdx.x * 16;

  // ---- stage A tile (16x128) as f16 + row sum-of-squares via ds_add_f32 ----
  if (tid < 16) shN1[tid] = 0.f;
  __syncthreads();
  {
    int row = tid >> 4;          // 0..15
    int k0 = (tid & 15) * 8;     // 0..120
    const float4* src = (const float4*)(desc1 + (size_t)(r0 + row) * DDIM + k0);
    float4 a = src[0], b = src[1];
    float ss = a.x * a.x + a.y * a.y + a.z * a.z + a.w * a.w +
               b.x * b.x + b.y * b.y + b.z * b.z + b.w * b.w;
    _Float16* dst = &At[row][k0];
    dst[0] = (_Float16)a.x; dst[1] = (_Float16)a.y;
    dst[2] = (_Float16)a.z; dst[3] = (_Float16)a.w;
    dst[4] = (_Float16)b.x; dst[5] = (_Float16)b.y;
    dst[6] = (_Float16)b.z; dst[7] = (_Float16)b.w;
    atomicAdd(&shN1[row], ss);
  }
  __syncthreads();

  // ---- loop-invariant A fragments (contiguous K per lane half, matches B, 4 K-chunks) ----
  // lane<16: halves 0..7 = K c*32+0..7,  halves 8..15 = K c*32+8..15
  // lane>=16: halves 0..7 = K c*32+16..23, halves 8..15 = K c*32+24..31
  const int arow = lane & 15;
  const int abase = (lane & 16) ? 16 : 0;
  v16h afrag[4];
#pragma unroll
  for (int c = 0; c < 4; c++) {
    v8h lo = *(const v8h*)&At[arow][c * 32 + abase];
    v8h hi = *(const v8h*)&At[arow][c * 32 + abase + 8];
    v16h a;
#pragma unroll
    for (int q = 0; q < 8; q++) { a[q] = lo[q]; a[q + 8] = hi[q]; }
    afrag[c] = a;
  }

  // per-thread running top-10 (registers)
  float tv[TOPK];
  int ti[TOPK];
#pragma unroll
  for (int k = 0; k < TOPK; k++) { tv[k] = FLT_BIG; ti[k] = 0; }
  float vmax = FLT_BIG;

  const int bcol = wave * 16 + (lane & 15);  // this lane's column in tile
  const int mbase = (lane & 16) ? 8 : 0;     // C layout: lanes 16-31 -> M=8+r
  const int srow = tid >> 4;                 // selection: row 0..15
  const int scol0 = (tid & 15) * 8;          // selection: 8 cols per thread

  // B-tile staging geometry (also used for next-tile prefetch)
  const int ldrow = tid >> 1;        // 0..127
  const int ldk0 = (tid & 1) * 64;   // 0 or 64

  for (int j0 = 0; j0 < B2; j0 += 128) {
    __syncthreads();  // everyone done with previous B tile / dist reads
    // ---- stage B tile: 128 cols x 128 K, f16, Bt[col][k] ----
    {
      const float4* src =
          (const float4*)(desc2 + (size_t)(j0 + ldrow) * DDIM + ldk0);
#pragma unroll
      for (int q = 0; q < 16; q++) {
        float4 v = src[q];
        _Float16* dst = &Bt[ldrow][ldk0 + q * 4];
        dst[0] = (_Float16)v.x; dst[1] = (_Float16)v.y;
        dst[2] = (_Float16)v.z; dst[3] = (_Float16)v.w;
      }
      // Pull next iteration's tile toward the WGP while we compute this one
      // (desc2 is L2-resident; this hides L2->WGP latency). Lowers to
      // global_prefetch_b8 on gfx1250.
      if (j0 + 128 < B2) {
        __builtin_prefetch(desc2 + (size_t)(j0 + 128 + ldrow) * DDIM + ldk0, 0,
                           1);
      }
    }
    __syncthreads();

    // ---- 16x16x128 via 4x v_wmma_f32_16x16x32_f16 ----
    v8f acc = {0.f, 0.f, 0.f, 0.f, 0.f, 0.f, 0.f, 0.f};
    {
      // B fragments, contiguous K as in A: lane<16 -> col=lane, K c*32+0..15;
      //                 lane>=16 -> col=lane-16, K c*32+16..31
      const int kb = (lane & 16) ? 16 : 0;
#pragma unroll
      for (int c = 0; c < 4; c++) {
        v16h b = *(const v16h*)&Bt[bcol][c * 32 + kb];
        acc = __builtin_amdgcn_wmma_f32_16x16x32_f16(
            false, afrag[c], false, b, (short)0, acc, false, false);
      }
    }
    // ---- squared distances into LDS tile ----
    {
      float n2v = n2g[j0 + bcol];
#pragma unroll
      for (int r = 0; r < 8; r++) {
        int M = mbase + r;
        dist[M][bcol] = shN1[M] + n2v - 2.0f * acc[r];
      }
    }
    __syncthreads();

    // ---- selection: each thread scans 8 cols of its row ----
    // Two ds_load_b128 instead of 8 scalar LDS reads; insert from registers.
    {
      float4 da = *(const float4*)&dist[srow][scol0];
      float4 db = *(const float4*)&dist[srow][scol0 + 4];
      float dv[8] = {da.x, da.y, da.z, da.w, db.x, db.y, db.z, db.w};
#pragma unroll
      for (int q = 0; q < 8; q++) {
        float d = dv[q];
        int j = j0 + scol0 + q;
        if (d < vmax) {
          bool done = false;
#pragma unroll
          for (int k = 0; k < TOPK; k++) {
            if (!done && tv[k] == vmax) { tv[k] = d; ti[k] = j; done = true; }
          }
          vmax = tv[0];
#pragma unroll
          for (int k = 1; k < TOPK; k++) vmax = fmaxf(vmax, tv[k]);
        }
      }
    }
  }

  // ---- merge 16 partial lists per row (merge bufs alias dead B tile) ----
  __syncthreads();
  {
    int row = tid >> 4, sub = tid & 15;
#pragma unroll
    for (int k = 0; k < TOPK; k++) {
      mvals[(row * 16 + sub) * TOPK + k] = tv[k];
      midx[(row * 16 + sub) * TOPK + k] = ti[k];
    }
  }
  __syncthreads();
  if (tid < 16) {
    int row = tid;
    float* mv = &mvals[row * 160];
    int* mi = &midx[row * 160];
    for (int k = 0; k < TOPK; k++) {
      float best = FLT_BIG;
      int bi = 0x7fffffff, bs = 0;
      for (int s = 0; s < 160; s++) {
        float v = mv[s];
        if (v < best || (v == best && mi[s] < bi)) { best = v; bi = mi[s]; bs = s; }
      }
      mv[bs] = FLT_BIG;
      valsOut[(size_t)(r0 + row) * TOPK + k] = sqrtf(fmaxf(best, 0.f));
      idxOut[(size_t)(r0 + row) * TOPK + k] = bi;
    }
  }
}

// ---------------- Kernel C: FGINN ratio test ----------------
__global__ void fginn_kernel(const float* __restrict__ vals,
                             const int* __restrict__ idxs,
                             const float* __restrict__ lafs2,
                             float* __restrict__ md, int* __restrict__ outIdx,
                             float* __restrict__ mask, int B1) {
  int i = blockIdx.x * blockDim.x + threadIdx.x;
  if (i >= B1) return;
  float v0 = vals[(size_t)i * TOPK];
  int best = idxs[(size_t)i * TOPK];
  float second = FLT_BIG;
#pragma unroll
  for (int k = 1; k < TOPK; k++) {
    int jk = idxs[(size_t)i * TOPK + k];
    int j0k = idxs[k];  // reference broadcasts ROW 0's k-th candidate
    float dx = lafs2[(size_t)jk * 6 + 2] - lafs2[(size_t)j0k * 6 + 2];
    float dy = lafs2[(size_t)jk * 6 + 5] - lafs2[(size_t)j0k * 6 + 5];
    float f = vals[(size_t)i * TOPK + k] +
              ((dx * dx + dy * dy) < 100.0f ? 1000000.0f : 0.0f);
    second = fminf(second, f);
  }
  float ratio = v0 / second;
  bool m = ratio <= 0.85f;
  md[i] = m ? ratio : 0.0f;
  outIdx[2 * i] = i;
  outIdx[2 * i + 1] = best;
  mask[i] = m ? 1.0f : 0.0f;
}

extern "C" void kernel_launch(void* const* d_in, const int* in_sizes, int n_in,
                              void* d_out, int out_size, void* d_ws,
                              size_t ws_size, hipStream_t stream) {
  const float* desc1 = (const float*)d_in[0];
  const float* desc2 = (const float*)d_in[1];
  const float* lafs2 = (const float*)d_in[3];  // (1,B2,2,3); centers at +2,+5
  int B1 = in_sizes[0] / DDIM;
  int B2 = in_sizes[1] / DDIM;

  // workspace: n2[B2] | vals[B1*10] | idxs[B1*10]  (~0.7 MB)
  float* n2 = (float*)d_ws;
  float* vals = n2 + B2;
  int* idxs = (int*)(vals + (size_t)B1 * TOPK);

  // d_out: match_dists[B1] f32 | matches_idxs[B1*2] i32 | mask[B1] (as f32)
  float* md = (float*)d_out;
  int* oi = (int*)(md + B1);
  float* mk = md + B1 + 2 * (size_t)B1;

  norms2_kernel<<<(B2 + 255) / 256, 256, 0, stream>>>(desc2, n2, B2);
  match_topk_kernel<<<B1 / 16, 256, 0, stream>>>(desc1, desc2, n2, vals, idxs,
                                                 B2);
  fginn_kernel<<<(B1 + 255) / 256, 256, 0, stream>>>(vals, idxs, lafs2, md, oi,
                                                     mk, B1);
}
